// WaveNet_73607149519326
// MI455X (gfx1250) — compile-verified, matrix operand order checked by reading, not run on hardware
//
#include <hip/hip_runtime.h>
#include <stdint.h>

#define DEVI __device__ __forceinline__

typedef __attribute__((ext_vector_type(16))) __bf16 v16bf;
typedef __attribute__((ext_vector_type(8)))  float  v8f;

static constexpr int T    = 16384;   // time steps
static constexpr int Hc   = 128;     // hidden channels
static constexpr int Sc   = 256;     // skip channels
static constexpr int MUc  = 256;     // output classes
static constexpr int MELP = 96;      // mel channels padded 80 -> 96
static constexpr int KK   = 512;     // embedding kernel
static constexpr int Lc   = 20;      // layers
static constexpr int TT   = 64;      // time tile per block
static constexpr int NB   = T / TT; // 256 blocks

// ---------------- async copy to LDS (gfx1250 ASYNCcnt path) ----------------

#if defined(__HIP_DEVICE_COMPILE__) && __has_builtin(__builtin_amdgcn_global_load_async_to_lds_b128)
#define HAVE_ASYNC_LDS 1
typedef int vint4 __attribute__((vector_size(16)));        // generic vector kind (matches builtin proto)
typedef __attribute__((address_space(1))) vint4 gvint4;    // __device__ (global)
typedef __attribute__((address_space(3))) vint4 lvint4;    // __shared__ (LDS)
#else
#define HAVE_ASYNC_LDS 0
#endif

// copy 16 bytes global -> LDS (per lane)
DEVI void async_cp16(const unsigned short* g, unsigned short* l) {
#if HAVE_ASYNC_LDS
  __builtin_amdgcn_global_load_async_to_lds_b128((gvint4*)g, (lvint4*)l, 0, 0);
#else
  *(uint4*)l = *(const uint4*)g;
#endif
}

DEVI void wait_async() {
#if HAVE_ASYNC_LDS
#if __has_builtin(__builtin_amdgcn_s_wait_asynccnt)
  __builtin_amdgcn_s_wait_asynccnt(0);
#else
  asm volatile("s_wait_asynccnt 0" ::: "memory");
#endif
#endif
}

// ---------------- helpers ----------------

DEVI unsigned short f2bf(float x) {
  unsigned u = __float_as_uint(x);
  u += 0x7FFFu + ((u >> 16) & 1u);   // round to nearest even
  return (unsigned short)(u >> 16);
}

DEVI int koff(int j, int hi) {       // ISA 16-bit A/B operand K-pair placement
  return (j < 4) ? (hi * 8 + 2 * j) : (16 + hi * 8 + 2 * (j - 4));
}

DEVI v8f wmma_bf16(v16bf a, v16bf b, v8f c) {
  return __builtin_amdgcn_wmma_f32_16x16x32_bf16(false, a, false, b,
                                                 (short)0, c, false, false);
}

// A fragment: 16(M) x 32(K) tile of row-major weights; W points at row 0 of
// the tile, row stride Ks elements (even -> coalesces to 2x b128 per lane).
DEVI v16bf afrag(const unsigned short* W, int Ks, int kb, int lane) {
  int m = lane & 15, hi = lane >> 4;
  const unsigned* p = (const unsigned*)(W + (size_t)m * Ks);
  union { unsigned u[8]; v16bf v; } r;
#pragma unroll
  for (int j = 0; j < 8; ++j) r.u[j] = p[(kb + koff(j, hi)) >> 1];
  return r.v;
}

// B fragment: 32(K) x 16(N) tile read from time-major [n][k] storage
// (LDS or global); X points at column 0 of the tile, row stride C (even).
DEVI v16bf bfrag(const unsigned short* X, int C, int kb, int lane) {
  int n = lane & 15, hi = lane >> 4;
  const unsigned* p = (const unsigned*)(X + (size_t)n * C);
  union { unsigned u[8]; v16bf v; } r;
#pragma unroll
  for (int j = 0; j < 8; ++j) r.u[j] = p[(kb + koff(j, hi)) >> 1];
  return r.v;
}

// B fragment for the Toeplitz wav window (value at wv[t_local + k]); pairs
// may be unaligned, so compose from 16-bit reads.
DEVI v16bf bfrag_wav(const unsigned short* wv, int base, int lane) {
  int n = lane & 15, hi = lane >> 4;
  union { unsigned short s[16]; v16bf v; } r;
#pragma unroll
  for (int j = 0; j < 8; ++j) {
    int idx = base + n + koff(j, hi);
    r.s[2 * j]     = wv[idx];
    r.s[2 * j + 1] = wv[idx + 1];
  }
  return r.v;
}

DEVI v8f vzero() { v8f z = {0.f, 0.f, 0.f, 0.f, 0.f, 0.f, 0.f, 0.f}; return z; }

// ---------------- prep kernels ----------------

__global__ __launch_bounds__(256) void k_cvt(const float* __restrict__ s,
                                             unsigned short* __restrict__ d, int n) {
  for (int i = blockIdx.x * 256 + threadIdx.x; i < n; i += gridDim.x * 256)
    d[i] = f2bf(s[i]);
}

// split Wf[l,o,i,2] into tap0 / tap1 bf16 matrices
__global__ __launch_bounds__(256) void k_cvt_pair(const float* __restrict__ s,
                                                  unsigned short* __restrict__ d0,
                                                  unsigned short* __restrict__ d1, int n) {
  for (int i = blockIdx.x * 256 + threadIdx.x; i < n; i += gridDim.x * 256) {
    d0[i] = f2bf(s[2 * i]);
    d1[i] = f2bf(s[2 * i + 1]);
  }
}

// pad rows of 80 to 96 (zero fill), convert to bf16
__global__ __launch_bounds__(256) void k_pad96(const float* __restrict__ s,
                                               unsigned short* __restrict__ d, int n) {
  for (int i = blockIdx.x * 256 + threadIdx.x; i < n; i += gridDim.x * 256) {
    int r = i / MELP, k = i % MELP;
    d[i] = (k < 80) ? f2bf(s[r * 80 + k]) : (unsigned short)0;
  }
}

__global__ __launch_bounds__(256) void k_addb(const float* __restrict__ a,
                                              const float* __restrict__ b,
                                              float* __restrict__ o, int n) {
  for (int i = blockIdx.x * 256 + threadIdx.x; i < n; i += gridDim.x * 256)
    o[i] = a[i] + b[i];
}

__global__ __launch_bounds__(256) void k_zero(float* __restrict__ p, int n) {
  for (int i = blockIdx.x * 256 + threadIdx.x; i < n; i += gridDim.x * 256)
    p[i] = 0.f;
}

// upsampled mel, time-major, padded to 96 channels, bf16
__global__ __launch_bounds__(256) void k_mel(const float* __restrict__ melspec,
                                             unsigned short* __restrict__ melT) {
  for (int i = blockIdx.x * 256 + threadIdx.x; i < T * MELP; i += gridDim.x * 256) {
    int t = i / MELP, k = i % MELP;
    unsigned short v = 0;
    if (k < 80) {
      int src = (int)(((long long)(t + 1) * 64) / (T + 1)); // nearest, after [:,:,1:]
      v = f2bf(melspec[k * 64 + src]);
    }
    melT[i] = v;
  }
}

// ---------------- embedding conv (K=512 causal, 1 -> 128ch) ----------------

__global__ __launch_bounds__(256) void emb_kernel(const float* __restrict__ wav,
                                                  const unsigned short* __restrict__ EW,
                                                  const float* __restrict__ eb,
                                                  float* __restrict__ h_out,
                                                  unsigned short* __restrict__ hb_out) {
  __shared__ unsigned short wv[TT + KK];   // wav[t0-511 .. t0+63] as bf16
  const int tid = threadIdx.x;
  const int lane = tid & 31, w = tid >> 5;
  const int t0 = blockIdx.x * TT;

  for (int i = tid; i < TT + KK; i += 256) {
    int g = t0 + i - (KK - 1);
    wv[i] = (g >= 0 && g < T) ? f2bf(wav[g]) : (unsigned short)0;
  }
  __syncthreads();

  const int orow = 16 * w;
  v8f C[4] = {vzero(), vzero(), vzero(), vzero()};
  for (int kb = 0; kb < KK; kb += 32) {
    v16bf a = afrag(EW + (size_t)orow * KK, KK, kb, lane);
#pragma unroll
    for (int c = 0; c < 4; ++c)
      C[c] = wmma_bf16(a, bfrag_wav(wv, 16 * c + kb, lane), C[c]);
  }

  const int hi = lane >> 4, n = lane & 15;
#pragma unroll
  for (int c = 0; c < 4; ++c)
#pragma unroll
    for (int r = 0; r < 8; ++r) {
      int o = orow + r + 8 * hi;
      int t = t0 + 16 * c + n;
      float v = C[c][r] + eb[o];
      h_out[(size_t)t * Hc + o]  = v;
      hb_out[(size_t)t * Hc + o] = f2bf(v);
    }
}

// ---------------- one WaveNet layer ----------------

__global__ __launch_bounds__(256) void layer_kernel(
    const float* __restrict__ h_in, float* __restrict__ h_out,
    const unsigned short* __restrict__ hb_in, unsigned short* __restrict__ hb_out,
    float* __restrict__ skip, const unsigned short* __restrict__ melT,
    const unsigned short* __restrict__ Wf0, const unsigned short* __restrict__ Wf1,
    const unsigned short* __restrict__ Wg0, const unsigned short* __restrict__ Wg1,
    const unsigned short* __restrict__ Vf,  const unsigned short* __restrict__ Vg,
    const unsigned short* __restrict__ SW,  const unsigned short* __restrict__ RW,
    const float* __restrict__ bf, const float* __restrict__ bg,
    const float* __restrict__ sb, const float* __restrict__ rb, int d) {
  __shared__ unsigned short hz_lds[TT * Hc];  // shifted h, then reused for z
  __shared__ unsigned short h_lds[TT * Hc];

  const int tid = threadIdx.x;
  const int lane = tid & 31, w = tid >> 5;
  const int t0 = blockIdx.x * TT;

  // ---- stage bf16 h tile via async LDS engine (16KB per tile) ----
  {
    const unsigned short* hsrc = hb_in + (size_t)t0 * Hc;
#pragma unroll
    for (int it = 0; it < 4; ++it) {
      int e = (it * 256 + tid) * 8;               // 8 ushorts = 16B per lane
      async_cp16(hsrc + e, h_lds + e);
    }
    if (t0 >= d) {                                // whole shifted tile in range
      const unsigned short* ssrc = hb_in + (size_t)(t0 - d) * Hc;
#pragma unroll
      for (int it = 0; it < 4; ++it) {
        int e = (it * 256 + tid) * 8;
        async_cp16(ssrc + e, hz_lds + e);
      }
    } else {                                      // head blocks: zero-padded
      for (int idx = tid; idx < TT * Hc; idx += 256) {
        int t = idx >> 7, c = idx & (Hc - 1);
        int ts = t0 + t - d;
        hz_lds[idx] = (ts >= 0) ? hb_in[(size_t)ts * Hc + c] : (unsigned short)0;
      }
    }
  }
  if (tid == 0) __builtin_prefetch(melT + (size_t)t0 * MELP, 0, 2);
  wait_async();
  __syncthreads();

  // ---- f/g pre-activations: Wf0@hs + Wf1@h + Vf@mel (and g-branch) ----
  const int orow = 16 * w;
  v8f Cf[4] = {vzero(), vzero(), vzero(), vzero()};
  v8f Cg[4] = {vzero(), vzero(), vzero(), vzero()};

  for (int kb = 0; kb < Hc; kb += 32) {                       // shifted h
    v16bf af = afrag(Wf0 + (size_t)orow * Hc, Hc, kb, lane);
    v16bf ag = afrag(Wg0 + (size_t)orow * Hc, Hc, kb, lane);
#pragma unroll
    for (int c = 0; c < 4; ++c) {
      v16bf b = bfrag(hz_lds + 16 * c * Hc, Hc, kb, lane);
      Cf[c] = wmma_bf16(af, b, Cf[c]);
      Cg[c] = wmma_bf16(ag, b, Cg[c]);
    }
  }
  for (int kb = 0; kb < Hc; kb += 32) {                       // current h
    v16bf af = afrag(Wf1 + (size_t)orow * Hc, Hc, kb, lane);
    v16bf ag = afrag(Wg1 + (size_t)orow * Hc, Hc, kb, lane);
#pragma unroll
    for (int c = 0; c < 4; ++c) {
      v16bf b = bfrag(h_lds + 16 * c * Hc, Hc, kb, lane);
      Cf[c] = wmma_bf16(af, b, Cf[c]);
      Cg[c] = wmma_bf16(ag, b, Cg[c]);
    }
  }
  for (int kb = 0; kb < MELP; kb += 32) {                     // mel conditioning
    v16bf af = afrag(Vf + (size_t)orow * MELP, MELP, kb, lane);
    v16bf ag = afrag(Vg + (size_t)orow * MELP, MELP, kb, lane);
#pragma unroll
    for (int c = 0; c < 4; ++c) {
      v16bf b = bfrag(melT + (size_t)(t0 + 16 * c) * MELP, MELP, kb, lane);
      Cf[c] = wmma_bf16(af, b, Cf[c]);
      Cg[c] = wmma_bf16(ag, b, Cg[c]);
    }
  }
  __syncthreads();  // everyone done reading hz_lds

  // ---- gated activation, write z (bf16) into reused LDS ----
  const int hi = lane >> 4, n = lane & 15;
#pragma unroll
  for (int c = 0; c < 4; ++c)
#pragma unroll
    for (int r = 0; r < 8; ++r) {
      int o = orow + r + 8 * hi;
      float fv = Cf[c][r] + bf[o];
      float gv = Cg[c][r] + bg[o];
      float zv = tanhf(fv) * (1.f / (1.f + __expf(-gv)));
      hz_lds[(16 * c + n) * Hc + o] = f2bf(zv);
    }
  __syncthreads();

  // ---- skip accumulation: skip += skip_w @ z + skip_b ----
#pragma unroll
  for (int rt = 0; rt < 2; ++rt) {
    int s0 = 16 * (2 * w + rt);
    v8f Cs[4] = {vzero(), vzero(), vzero(), vzero()};
    for (int kb = 0; kb < Hc; kb += 32) {
      v16bf a = afrag(SW + (size_t)s0 * Hc, Hc, kb, lane);
#pragma unroll
      for (int c = 0; c < 4; ++c)
        Cs[c] = wmma_bf16(a, bfrag(hz_lds + 16 * c * Hc, Hc, kb, lane), Cs[c]);
    }
#pragma unroll
    for (int c = 0; c < 4; ++c)
#pragma unroll
      for (int r = 0; r < 8; ++r) {
        int s = s0 + r + 8 * hi;
        int t = t0 + 16 * c + n;
        size_t gi = (size_t)t * Sc + s;
        skip[gi] = skip[gi] + Cs[c][r] + sb[s];
      }
  }

  // ---- residual: h_out = res_w @ z + res_b + h_in (f32 + bf16 shadow) ----
  v8f Cr[4] = {vzero(), vzero(), vzero(), vzero()};
  for (int kb = 0; kb < Hc; kb += 32) {
    v16bf a = afrag(RW + (size_t)orow * Hc, Hc, kb, lane);
#pragma unroll
    for (int c = 0; c < 4; ++c)
      Cr[c] = wmma_bf16(a, bfrag(hz_lds + 16 * c * Hc, Hc, kb, lane), Cr[c]);
  }
#pragma unroll
  for (int c = 0; c < 4; ++c)
#pragma unroll
    for (int r = 0; r < 8; ++r) {
      int o = orow + r + 8 * hi;
      int t = t0 + 16 * c + n;
      size_t gi = (size_t)t * Hc + o;
      float v = Cr[c][r] + rb[o] + h_in[gi];
      h_out[gi]  = v;
      hb_out[gi] = f2bf(v);
    }
}

// ---------------- output stage 1: v = relu(out_w @ relu(skip) + out_b) ------

__global__ __launch_bounds__(256) void dense1(const float* __restrict__ skip,
                                              const unsigned short* __restrict__ OW,
                                              const float* __restrict__ ob,
                                              unsigned short* __restrict__ vbuf) {
  __shared__ unsigned short xb[TT * Sc];   // 32 KB
  const int tid = threadIdx.x;
  const int lane = tid & 31, w = tid >> 5;
  const int t0 = blockIdx.x * TT;

  for (int idx = tid; idx < TT * Sc; idx += 256) {
    int t = idx >> 8, c = idx & (Sc - 1);
    xb[idx] = f2bf(fmaxf(skip[(size_t)(t0 + t) * Sc + c], 0.f));
  }
  __syncthreads();

  const int hi = lane >> 4, n = lane & 15;
#pragma unroll
  for (int rt = 0; rt < 2; ++rt) {
    int o0 = 16 * (2 * w + rt);
    v8f C[4] = {vzero(), vzero(), vzero(), vzero()};
    for (int kb = 0; kb < Sc; kb += 32) {
      v16bf a = afrag(OW + (size_t)o0 * Sc, Sc, kb, lane);
#pragma unroll
      for (int c = 0; c < 4; ++c)
        C[c] = wmma_bf16(a, bfrag(xb + 16 * c * Sc, Sc, kb, lane), C[c]);
    }
#pragma unroll
    for (int c = 0; c < 4; ++c)
#pragma unroll
      for (int r = 0; r < 8; ++r) {
        int o = o0 + r + 8 * hi;
        int t = t0 + 16 * c + n;
        vbuf[(size_t)t * MUc + o] = f2bf(fmaxf(C[c][r] + ob[o], 0.f));
      }
  }
}

// ---------------- output stage 2: out[o][t] = end_w @ v + end_b -------------

__global__ __launch_bounds__(256) void dense2(const unsigned short* __restrict__ vbuf,
                                              const unsigned short* __restrict__ EWn,
                                              const float* __restrict__ eb,
                                              float* __restrict__ outp) {
  __shared__ unsigned short xb[TT * MUc];
  const int tid = threadIdx.x;
  const int lane = tid & 31, w = tid >> 5;
  const int t0 = blockIdx.x * TT;

  {
    const unsigned short* src = vbuf + (size_t)t0 * MUc;
#pragma unroll
    for (int it = 0; it < 8; ++it) {            // 64*256 ushorts = 32KB
      int e = (it * 256 + tid) * 8;
      async_cp16(src + e, xb + e);
    }
  }
  wait_async();
  __syncthreads();

  const int hi = lane >> 4, n = lane & 15;
#pragma unroll
  for (int rt = 0; rt < 2; ++rt) {
    int o0 = 16 * (2 * w + rt);
    v8f C[4] = {vzero(), vzero(), vzero(), vzero()};
    for (int kb = 0; kb < MUc; kb += 32) {
      v16bf a = afrag(EWn + (size_t)o0 * MUc, MUc, kb, lane);
#pragma unroll
      for (int c = 0; c < 4; ++c)
        C[c] = wmma_bf16(a, bfrag(xb + 16 * c * MUc, MUc, kb, lane), C[c]);
    }
#pragma unroll
    for (int c = 0; c < 4; ++c)
#pragma unroll
      for (int r = 0; r < 8; ++r) {
        int o = o0 + r + 8 * hi;
        int t = t0 + 16 * c + n;
        outp[(size_t)o * T + t] = C[c][r] + eb[o];   // [MU][T] channel-major
      }
  }
}

// ---------------- host launch ----------------

extern "C" void kernel_launch(void* const* d_in, const int* in_sizes, int n_in,
                              void* d_out, int out_size, void* d_ws, size_t ws_size,
                              hipStream_t stream) {
  (void)in_sizes; (void)n_in; (void)out_size; (void)ws_size;
  const float* melspec = (const float*)d_in[0];
  const float* wav     = (const float*)d_in[1];
  const float* emb_w   = (const float*)d_in[2];
  const float* emb_b   = (const float*)d_in[3];
  const float* Wf      = (const float*)d_in[4];
  const float* Wf_b    = (const float*)d_in[5];
  const float* Wg      = (const float*)d_in[6];
  const float* Wg_b    = (const float*)d_in[7];
  const float* Vf      = (const float*)d_in[8];
  const float* Vf_b    = (const float*)d_in[9];
  const float* Vg      = (const float*)d_in[10];
  const float* Vg_b    = (const float*)d_in[11];
  const float* skip_w  = (const float*)d_in[12];
  const float* skip_b  = (const float*)d_in[13];
  const float* res_w   = (const float*)d_in[14];
  const float* res_b   = (const float*)d_in[15];
  const float* out_w   = (const float*)d_in[16];
  const float* out_b   = (const float*)d_in[17];
  const float* end_w   = (const float*)d_in[18];
  const float* end_b   = (const float*)d_in[19];
  float* outp = (float*)d_out;

  char* ws = (char*)d_ws;
  size_t off = 0;
  auto alloc = [&](size_t bytes) -> char* {
    char* p = ws + off;
    off = (off + bytes + 255) & ~(size_t)255;
    return p;
  };

  float*          hA    = (float*)alloc((size_t)T * Hc * 4);
  float*          hB    = (float*)alloc((size_t)T * Hc * 4);
  unsigned short* hbA   = (unsigned short*)alloc((size_t)T * Hc * 2);
  unsigned short* hbB   = (unsigned short*)alloc((size_t)T * Hc * 2);
  float*          skip  = (float*)alloc((size_t)T * Sc * 4);
  unsigned short* melT  = (unsigned short*)alloc((size_t)T * MELP * 2);
  unsigned short* vbuf  = (unsigned short*)alloc((size_t)T * MUc * 2);
  unsigned short* Wf0b  = (unsigned short*)alloc((size_t)Lc * Hc * Hc * 2);
  unsigned short* Wf1b  = (unsigned short*)alloc((size_t)Lc * Hc * Hc * 2);
  unsigned short* Wg0b  = (unsigned short*)alloc((size_t)Lc * Hc * Hc * 2);
  unsigned short* Wg1b  = (unsigned short*)alloc((size_t)Lc * Hc * Hc * 2);
  unsigned short* Vfb   = (unsigned short*)alloc((size_t)Lc * Hc * MELP * 2);
  unsigned short* Vgb   = (unsigned short*)alloc((size_t)Lc * Hc * MELP * 2);
  unsigned short* SWb   = (unsigned short*)alloc((size_t)Lc * Sc * Hc * 2);
  unsigned short* RWb   = (unsigned short*)alloc((size_t)Lc * Hc * Hc * 2);
  unsigned short* EWb   = (unsigned short*)alloc((size_t)Hc * KK * 2);
  unsigned short* OWb   = (unsigned short*)alloc((size_t)MUc * Sc * 2);
  unsigned short* ENDb  = (unsigned short*)alloc((size_t)MUc * MUc * 2);
  float*          biasf = (float*)alloc((size_t)Lc * Hc * 4);
  float*          biasg = (float*)alloc((size_t)Lc * Hc * 4);

  auto gsz = [](int n) { return dim3((n + 255) / 256); };

  // ---- prep ----
  k_mel<<<gsz(T * MELP), 256, 0, stream>>>(melspec, melT);
  k_cvt<<<gsz(Hc * KK), 256, 0, stream>>>(emb_w, EWb, Hc * KK);
  k_cvt<<<gsz(MUc * Sc), 256, 0, stream>>>(out_w, OWb, MUc * Sc);
  k_cvt<<<gsz(MUc * MUc), 256, 0, stream>>>(end_w, ENDb, MUc * MUc);
  k_cvt<<<gsz(Lc * Sc * Hc), 256, 0, stream>>>(skip_w, SWb, Lc * Sc * Hc);
  k_cvt<<<gsz(Lc * Hc * Hc), 256, 0, stream>>>(res_w, RWb, Lc * Hc * Hc);
  k_cvt_pair<<<gsz(Lc * Hc * Hc), 256, 0, stream>>>(Wf, Wf0b, Wf1b, Lc * Hc * Hc);
  k_cvt_pair<<<gsz(Lc * Hc * Hc), 256, 0, stream>>>(Wg, Wg0b, Wg1b, Lc * Hc * Hc);
  k_pad96<<<gsz(Lc * Hc * MELP), 256, 0, stream>>>(Vf, Vfb, Lc * Hc * MELP);
  k_pad96<<<gsz(Lc * Hc * MELP), 256, 0, stream>>>(Vg, Vgb, Lc * Hc * MELP);
  k_addb<<<gsz(Lc * Hc), 256, 0, stream>>>(Wf_b, Vf_b, biasf, Lc * Hc);
  k_addb<<<gsz(Lc * Hc), 256, 0, stream>>>(Wg_b, Vg_b, biasg, Lc * Hc);
  k_zero<<<gsz(T * Sc), 256, 0, stream>>>(skip, T * Sc);

  // ---- embedding conv ----
  emb_kernel<<<NB, 256, 0, stream>>>(wav, EWb, emb_b, hA, hbA);

  // ---- 20 gated layers (ping-pong h) ----
  const float*          hin  = hA;
  float*                hout = hB;
  const unsigned short* hbin = hbA;
  unsigned short*       hbout = hbB;
  for (int l = 0; l < Lc; ++l) {
    int d = 1 << (l % 10);
    layer_kernel<<<NB, 256, 0, stream>>>(
        hin, hout, hbin, hbout, skip, melT,
        Wf0b + (size_t)l * Hc * Hc, Wf1b + (size_t)l * Hc * Hc,
        Wg0b + (size_t)l * Hc * Hc, Wg1b + (size_t)l * Hc * Hc,
        Vfb + (size_t)l * Hc * MELP, Vgb + (size_t)l * Hc * MELP,
        SWb + (size_t)l * Sc * Hc, RWb + (size_t)l * Hc * Hc,
        biasf + (size_t)l * Hc, biasg + (size_t)l * Hc,
        skip_b + (size_t)l * Sc, res_b + (size_t)l * Hc, d);
    float* tf = (float*)hin; hin = hout; hout = tf;
    unsigned short* tb = (unsigned short*)hbin; hbin = hbout; hbout = tb;
  }

  // ---- output head ----
  dense1<<<NB, 256, 0, stream>>>(skip, OWb, out_b, vbuf);
  dense2<<<NB, 256, 0, stream>>>(vbuf, ENDb, end_b, outp);
}
